// TidalAttention_83202106458557
// MI455X (gfx1250) — matrix operand order repaired and checked by reading, not compiled, not run on hardware
//
#include <hip/hip_runtime.h>
#include <hip/hip_bf16.h>

// ---------------------------------------------------------------------------
// TidalAttention forward for MI455X (gfx1250, wave32, WMMA + Tensor Data Mover)
//
//   B=4, S=2048, H=16, DH=64, DM=1024, fp32 in / fp32 out.
//   flip->softmax->flip == softmax (permutation invariant) -> flash softmax.
//
// Pipeline:
//   1) cvt:   X, Wq/Wk/Wv  fp32 -> f16 in workspace
//   2) qkv:   Q/K = X@W^T + b  stored [b,h,s,d] f16 ; V stored [b,h,d,s] f16
//             X/W k-tiles staged in LDS by TDM (double buffered)
//   3) attn:  per-wave 16 q-rows online-softmax; K/V tiles staged by TDM
// ---------------------------------------------------------------------------

typedef _Float16 v16h __attribute__((ext_vector_type(16)));
typedef _Float16 v8h  __attribute__((ext_vector_type(8)));
typedef _Float16 v4h  __attribute__((ext_vector_type(4)));
typedef float    v8f  __attribute__((ext_vector_type(8)));
typedef unsigned int v4u __attribute__((ext_vector_type(4)));
typedef int      v8i  __attribute__((ext_vector_type(8)));
typedef int      v4i  __attribute__((ext_vector_type(4)));

#define NB   4
#define NH   16
#define SEQ  2048
#define DHD  64
#define DMDL 1024

union Frag16 {
    v16h v;
    v8h  h[2];
    _Float16 e[16];
};

static __device__ __forceinline__ v8f wmma_f16(const Frag16& a, const Frag16& b, v8f c) {
    return __builtin_amdgcn_wmma_f32_16x16x32_f16(false, a.v, false, b.v,
                                                  (short)0, c, false, false);
}

// LDS byte offset of a __shared__ object: low 32 bits of its flat address
static __device__ __forceinline__ unsigned lds_off(const void* p) {
    return (unsigned)(unsigned long long)p;
}

// ---------------------------------------------------------------------------
// Tensor Data Mover: 2D tile load, global -> LDS.  Elements are 2 bytes (f16).
// D# per CDNA5 ISA ch.8: group0 = {count, lds_addr, global_addr, type=2},
// group1 = {data_size, tensor dims, tile dims, dim0 stride}.
// Issued by one wave (EXEC ignored); tracked by TENSORcnt.
// ---------------------------------------------------------------------------
static __device__ __forceinline__ void tdm_load_2d_f16(
    unsigned lds_addr, const void* gaddr,
    unsigned tensor_d0, unsigned tensor_d1,
    unsigned tile_d0, unsigned tile_d1, unsigned stride0_elems)
{
    unsigned long long ga = (unsigned long long)gaddr;
    v4u g0;
    g0[0] = 1u;                                   // count=1, user descriptor
    g0[1] = lds_addr;                             // LDS byte address
    g0[2] = (unsigned)ga;                         // global_addr[31:0]
    g0[3] = (unsigned)((ga >> 32) & 0x1FFFFFFu)   // global_addr[56:32]
          | (2u << 30);                           // type = 2 ("image")
    v8i g1;
    g1[0] = (int)(1u << 16);                      // data_size=1 -> 2 bytes
    g1[1] = (int)((tensor_d0 & 0xFFFFu) << 16);   // tensor_dim0[15:0] @ bits 63:48
    g1[2] = (int)((tensor_d0 >> 16) | ((tensor_d1 & 0xFFFFu) << 16));
    g1[3] = (int)((tensor_d1 >> 16) | (tile_d0 << 16));
    g1[4] = (int)(tile_d1);                       // tile_dim1 ; tile_dim2 = 0
    g1[5] = (int)(stride0_elems);                 // tensor_dim0_stride[31:0]
    g1[6] = 0;
    g1[7] = 0;
    v4i z4 = {0, 0, 0, 0};
    v8i z8 = {0, 0, 0, 0, 0, 0, 0, 0};
    __builtin_amdgcn_tensor_load_to_lds(g0, g1, z4, z4, z8, 0);
}

// ---------------------------------------------------------------------------
// Kernel 1: fp32 -> f16 conversion (n multiple of 4)
// ---------------------------------------------------------------------------
__global__ void tidal_cvt_f16(const float* __restrict__ src,
                              _Float16* __restrict__ dst, int n) {
    int i = (blockIdx.x * blockDim.x + threadIdx.x) * 4;
    if (i < n) {
        float4 f = *(const float4*)(src + i);
        v4h h;
        h[0] = (_Float16)f.x; h[1] = (_Float16)f.y;
        h[2] = (_Float16)f.z; h[3] = (_Float16)f.w;
        *(v4h*)(dst + i) = h;
    }
}

// ---------------------------------------------------------------------------
// Kernel 2: fused QKV projection.  grid = (N/64, M/128, 3), block = 128 (4 waves)
//   Y = Xh @ W^T + bias   (M=8192, N=1024, K=1024)
//   Block tile 128x64.  X (128x32) and W (64x32) k-tiles staged by TDM,
//   double buffered.  Each wave: 32 rows x 64 cols = 2 A-frags x 4 B-frags
//   = 8 WMMAs per k-step over 8 f32 accumulators.
//   z=0 -> Q [b,h,s,d] ; z=1 -> K [b,h,s,d] ; z=2 -> V transposed [b,h,d,s]
// ---------------------------------------------------------------------------
__global__ __launch_bounds__(128)
void tidal_qkv_gemm(const _Float16* __restrict__ Xh,
                    const _Float16* __restrict__ WhAll,
                    const float* __restrict__ bq,
                    const float* __restrict__ bk,
                    const float* __restrict__ bv,
                    _Float16* __restrict__ Qh,
                    _Float16* __restrict__ Kh,
                    _Float16* __restrict__ Vt) {
    __shared__ __align__(16) _Float16 xs[2][128][32];  // [buf][row][k]
    __shared__ __align__(16) _Float16 wsb[2][64][32];  // [buf][col][k]

    const int wave = threadIdx.x >> 5;
    const int lane = threadIdx.x & 31;
    const int hf   = lane >> 4;
    const int lr   = lane & 15;

    const int z  = blockIdx.z;
    const int m0 = blockIdx.y * 128;
    const int n0 = blockIdx.x * 64;

    const _Float16* __restrict__ W    = WhAll + (size_t)z * (DMDL * DMDL);
    const float*    __restrict__ bias = (z == 0) ? bq : (z == 1) ? bk : bv;

    const _Float16* Xtile = Xh + (size_t)m0 * DMDL;   // 128 rows, stride DMDL
    const _Float16* Wtile = W  + (size_t)n0 * DMDL;   // 64 rows, stride DMDL

    v8f acc[2][4];
#pragma unroll
    for (int rt = 0; rt < 2; rt++)
#pragma unroll
        for (int nt = 0; nt < 4; nt++) acc[rt][nt] = v8f{0};

    // Preload k=0 tiles via TDM (wave 0 issues; TDM ignores EXEC)
    if (wave == 0) {
        tdm_load_2d_f16(lds_off(&xs[0][0][0]),  Xtile, 32, 128, 32, 128, DMDL);
        tdm_load_2d_f16(lds_off(&wsb[0][0][0]), Wtile, 32, 64,  32, 64,  DMDL);
    }

    int cur = 0;
    for (int k = 0; k < DMDL; k += 32) {
        if (wave == 0) __builtin_amdgcn_s_wait_tensorcnt((short)0);
        __syncthreads();  // current tiles ready; previous reads finished

        if (wave == 0 && (k + 32) < DMDL) {  // prefetch next k-tile into other buf
            tdm_load_2d_f16(lds_off(&xs[cur ^ 1][0][0]),  Xtile + k + 32,
                            32, 128, 32, 128, DMDL);
            tdm_load_2d_f16(lds_off(&wsb[cur ^ 1][0][0]), Wtile + k + 32,
                            32, 64,  32, 64,  DMDL);
        }

        // Load ALL fragments first (one LDS clause), then back-to-back WMMAs.
        Frag16 a[2], bf[4];
#pragma unroll
        for (int rt = 0; rt < 2; rt++) {
            const int row = wave * 32 + rt * 16 + lr;
            a[rt].h[0] = *(const v8h*)&xs[cur][row][hf * 8];
            a[rt].h[1] = *(const v8h*)&xs[cur][row][16 + hf * 8];
        }
#pragma unroll
        for (int nt = 0; nt < 4; nt++) {
            bf[nt].h[0] = *(const v8h*)&wsb[cur][nt * 16 + lr][hf * 8];
            bf[nt].h[1] = *(const v8h*)&wsb[cur][nt * 16 + lr][16 + hf * 8];
        }
#pragma unroll
        for (int rt = 0; rt < 2; rt++)
#pragma unroll
            for (int nt = 0; nt < 4; nt++)
                acc[rt][nt] = wmma_f16(a[rt], bf[nt], acc[rt][nt]);
        cur ^= 1;
    }

#pragma unroll
    for (int rt = 0; rt < 2; rt++) {
#pragma unroll
        for (int nt = 0; nt < 4; nt++) {
            const int n  = n0 + nt * 16 + lr;        // output column
            const int hh = n >> 6;
            const int d  = n & 63;
            const float bv_ = bias[n];
            const int m  = m0 + wave * 32 + rt * 16 + 8 * hf;  // first row
            const int bb = m >> 11;
            const int s  = m & 2047;
            if (z < 2) {
                _Float16* __restrict__ Out = (z == 0) ? Qh : Kh;
                size_t base = ((size_t)(bb * NH + hh) * SEQ + s) * DHD + d;
#pragma unroll
                for (int i = 0; i < 8; i++)
                    Out[base + (size_t)i * DHD] = (_Float16)(acc[rt][nt][i] + bv_);
            } else {
                v8h pk;
#pragma unroll
                for (int i = 0; i < 8; i++) pk[i] = (_Float16)(acc[rt][nt][i] + bv_);
                *(v8h*)(Vt + ((size_t)(bb * NH + hh) * DHD + d) * SEQ + s) = pk;
            }
        }
    }
}

// ---------------------------------------------------------------------------
// Kernel 3: attention.  grid = (S/64, H, B), block = 128 (4 waves).
// K tile (32 keys x 64 d) and V tile (64 d x 32 keys, from V^T) staged in LDS
// by TDM, double buffered; each wave owns 16 query rows, online softmax.
// ---------------------------------------------------------------------------
__global__ __launch_bounds__(128)
void tidal_attn(const _Float16* __restrict__ Qh,
                const _Float16* __restrict__ Kh,
                const _Float16* __restrict__ Vt,
                const float* __restrict__ mask,
                float* __restrict__ out) {
    __shared__ __align__(16) _Float16 kbuf[2][32][64];  // [buf][key][d]
    __shared__ __align__(16) _Float16 vbuf[2][64][32];  // [buf][d][key]
    __shared__ __align__(16) _Float16 plds[4][16][36];  // per-wave P relayout

    const int wave = threadIdx.x >> 5;
    const int lane = threadIdx.x & 31;
    const int hf   = lane >> 4;
    const int lr   = lane & 15;

    const int b  = blockIdx.z;
    const int h  = blockIdx.y;
    const int q0 = blockIdx.x * 64 + wave * 16;

    const _Float16* __restrict__ Qb = Qh + (size_t)(b * NH + h) * SEQ * DHD;
    const _Float16* __restrict__ Kb = Kh + (size_t)(b * NH + h) * SEQ * DHD;
    const _Float16* __restrict__ Vb = Vt + (size_t)(b * NH + h) * DHD * SEQ;
    const float*    __restrict__ mb = mask + (size_t)b * SEQ;

    // Q A-fragments for the full DH=64 contraction (two 16x32 fragments)
    const _Float16* Qrow = Qb + (size_t)(q0 + lr) * DHD;
    Frag16 aq0, aq1;
    aq0.h[0] = *(const v8h*)(Qrow + hf * 8);
    aq0.h[1] = *(const v8h*)(Qrow + 16 + hf * 8);
    aq1.h[0] = *(const v8h*)(Qrow + 32 + hf * 8);
    aq1.h[1] = *(const v8h*)(Qrow + 48 + hf * 8);

    v8f o[4] = {v8f{0}, v8f{0}, v8f{0}, v8f{0}};
    float mrow[8], lrow[8];
#pragma unroll
    for (int i = 0; i < 8; i++) { mrow[i] = -1e30f; lrow[i] = 0.0f; }

    const float scale = 0.125f;  // 1/sqrt(DH)

    // Preload first K/V tiles via TDM
    if (wave == 0) {
        tdm_load_2d_f16(lds_off(&kbuf[0][0][0]), Kb, DHD, SEQ, DHD, 32, DHD);
        tdm_load_2d_f16(lds_off(&vbuf[0][0][0]), Vb, SEQ, DHD, 32, DHD, SEQ);
    }

    int cur = 0;
    for (int sk = 0; sk < SEQ; sk += 32) {
        if (wave == 0) __builtin_amdgcn_s_wait_tensorcnt((short)0);
        __syncthreads();

        if (wave == 0 && (sk + 32) < SEQ) {  // prefetch next key tile
            tdm_load_2d_f16(lds_off(&kbuf[cur ^ 1][0][0]),
                            Kb + (size_t)(sk + 32) * DHD, DHD, SEQ, DHD, 32, DHD);
            tdm_load_2d_f16(lds_off(&vbuf[cur ^ 1][0][0]),
                            Vb + sk + 32, SEQ, DHD, 32, DHD, SEQ);
        }

        // ---- scores: 16 q-rows x 32 keys, contraction over d=64 ----
        // Load all 4 B-fragments first, then 4 WMMAs back-to-back.
        Frag16 bk0[2], bk1[2];
#pragma unroll
        for (int kt = 0; kt < 2; kt++) {
            bk0[kt].h[0] = *(const v8h*)&kbuf[cur][kt * 16 + lr][hf * 8];
            bk0[kt].h[1] = *(const v8h*)&kbuf[cur][kt * 16 + lr][16 + hf * 8];
            bk1[kt].h[0] = *(const v8h*)&kbuf[cur][kt * 16 + lr][32 + hf * 8];
            bk1[kt].h[1] = *(const v8h*)&kbuf[cur][kt * 16 + lr][48 + hf * 8];
        }
        v8f sc[2] = {v8f{0}, v8f{0}};
#pragma unroll
        for (int kt = 0; kt < 2; kt++) {
            sc[kt] = wmma_f16(aq0, bk0[kt], sc[kt]);
            sc[kt] = wmma_f16(aq1, bk1[kt], sc[kt]);
        }

        const float mk0 = mb[sk + lr];
        const float mk1 = mb[sk + 16 + lr];

        float tmax[8];
#pragma unroll
        for (int i = 0; i < 8; i++) {
            sc[0][i] = sc[0][i] * scale + mk0;
            sc[1][i] = sc[1][i] * scale + mk1;
            tmax[i]  = fmaxf(sc[0][i], sc[1][i]);
        }
#pragma unroll
        for (int off = 1; off < 16; off <<= 1)
#pragma unroll
            for (int i = 0; i < 8; i++)
                tmax[i] = fmaxf(tmax[i], __shfl_xor(tmax[i], off, 16));

        float ex[8], psum[8];
#pragma unroll
        for (int i = 0; i < 8; i++) {
            float mn = fmaxf(mrow[i], tmax[i]);
            ex[i]    = __expf(mrow[i] - mn);
            mrow[i]  = mn;
            sc[0][i] = __expf(sc[0][i] - mn);
            sc[1][i] = __expf(sc[1][i] - mn);
            psum[i]  = sc[0][i] + sc[1][i];
        }
#pragma unroll
        for (int off = 1; off < 16; off <<= 1)
#pragma unroll
            for (int i = 0; i < 8; i++)
                psum[i] += __shfl_xor(psum[i], off, 16);
#pragma unroll
        for (int i = 0; i < 8; i++) lrow[i] = lrow[i] * ex[i] + psum[i];
#pragma unroll
        for (int t = 0; t < 4; t++)
#pragma unroll
            for (int i = 0; i < 8; i++) o[t][i] *= ex[i];

        // ---- relayout P (C-layout) -> A-fragment via per-wave LDS ----
#pragma unroll
        for (int i = 0; i < 8; i++) {
            plds[wave][i + 8 * hf][lr]      = (_Float16)sc[0][i];
            plds[wave][i + 8 * hf][16 + lr] = (_Float16)sc[1][i];
        }
        __builtin_amdgcn_wave_barrier();  // same-wave LDS ops are in-order
        Frag16 pa;
#pragma unroll
        for (int j = 0; j < 8; j++) {
            pa.e[j]     = plds[wave][lr][hf * 8 + j];
            pa.e[8 + j] = plds[wave][lr][16 + hf * 8 + j];
        }
        __builtin_amdgcn_wave_barrier();

        // ---- ctx accumulate: o += P(16x32) @ V(32x64) ----
        Frag16 bvf[4];
#pragma unroll
        for (int t = 0; t < 4; t++) {
            bvf[t].h[0] = *(const v8h*)&vbuf[cur][t * 16 + lr][hf * 8];
            bvf[t].h[1] = *(const v8h*)&vbuf[cur][t * 16 + lr][16 + hf * 8];
        }
#pragma unroll
        for (int t = 0; t < 4; t++) o[t] = wmma_f16(pa, bvf[t], o[t]);
        cur ^= 1;
    }

    // ---- normalize and store [B,S,DM] fp32 ----
    float inv[8];
#pragma unroll
    for (int i = 0; i < 8; i++) inv[i] = 1.0f / lrow[i];
#pragma unroll
    for (int t = 0; t < 4; t++) {
        const int d = h * DHD + t * 16 + lr;
#pragma unroll
        for (int i = 0; i < 8; i++) {
            const int s = q0 + i + 8 * hf;
            out[((size_t)(b * SEQ + s)) * DMDL + d] = o[t][i] * inv[i];
        }
    }
}

// ---------------------------------------------------------------------------
// Host launcher
// ---------------------------------------------------------------------------
extern "C" void kernel_launch(void* const* d_in, const int* in_sizes, int n_in,
                              void* d_out, int out_size, void* d_ws, size_t ws_size,
                              hipStream_t stream) {
    const float* X    = (const float*)d_in[0];
    const float* mask = (const float*)d_in[1];
    const float* Wq   = (const float*)d_in[2];
    const float* bq   = (const float*)d_in[3];
    const float* Wk   = (const float*)d_in[4];
    const float* bk   = (const float*)d_in[5];
    const float* Wv   = (const float*)d_in[6];
    const float* bv   = (const float*)d_in[7];
    float* out = (float*)d_out;

    const int XN = NB * SEQ * DMDL;      // 8388608
    const int WN = DMDL * DMDL;          // 1048576
    const int QN = NB * NH * SEQ * DHD;  // 8388608

    _Float16* base = (_Float16*)d_ws;
    _Float16* Xh = base;                 // f16 X
    _Float16* Wh = Xh + XN;              // f16 Wq,Wk,Wv (contiguous)
    _Float16* Qh = Wh + 3 * WN;          // Q [b,h,s,d]
    _Float16* Kh = Qh + QN;              // K [b,h,s,d]
    _Float16* Vt = Kh + QN;              // V [b,h,d,s]

    tidal_cvt_f16<<<XN / 4 / 256, 256, 0, stream>>>(X, Xh, XN);
    tidal_cvt_f16<<<WN / 4 / 256, 256, 0, stream>>>(Wq, Wh + 0 * WN, WN);
    tidal_cvt_f16<<<WN / 4 / 256, 256, 0, stream>>>(Wk, Wh + 1 * WN, WN);
    tidal_cvt_f16<<<WN / 4 / 256, 256, 0, stream>>>(Wv, Wh + 2 * WN, WN);

    tidal_qkv_gemm<<<dim3(DMDL / 64, (NB * SEQ) / 128, 3), 128, 0, stream>>>(
        Xh, Wh, bq, bk, bv, Qh, Kh, Vt);

    tidal_attn<<<dim3(SEQ / 64, NH, NB), 128, 0, stream>>>(Qh, Kh, Vt, mask, out);
}
